// Net_64716567216583
// MI455X (gfx1250) — matrix operand order repaired and checked by reading, not compiled, not run on hardware
//
#include <hip/hip_runtime.h>
#include <stdint.h>

#define B_ 256
#define T_ 100
#define I_ 128
#define H_ 2048

typedef __attribute__((ext_vector_type(16))) __bf16 v16bf;
typedef __attribute__((ext_vector_type(8)))  __bf16 v8bf;
typedef __attribute__((ext_vector_type(8)))  float  v8f;
typedef __attribute__((ext_vector_type(4)))  int    v4i;

// address-space qualified pointee types for the async-LDS builtin
typedef __attribute__((address_space(1))) v4i v4i_as1;
typedef __attribute__((address_space(3))) v4i v4i_as3;

// ---- gfx1250 async global->LDS copy (ASYNCcnt path), with safe fallback ----
#if __has_builtin(__builtin_amdgcn_global_load_async_to_lds_b128)
#define HAVE_ASYNC_LDS 1
__device__ __forceinline__ void async_copy16(const unsigned short* g, unsigned short* l) {
    __builtin_amdgcn_global_load_async_to_lds_b128((v4i_as1*)g, (v4i_as3*)l, 0, 0);
}
#else
#define HAVE_ASYNC_LDS 0
__device__ __forceinline__ void async_copy16(const unsigned short* g, unsigned short* l) {
    *(uint4*)l = *(const uint4*)g;
}
#endif

#if __has_builtin(__builtin_amdgcn_s_wait_asynccnt)
#define WAIT_ASYNC(N) __builtin_amdgcn_s_wait_asynccnt(N)
#elif HAVE_ASYNC_LDS
#define WAIT_ASYNC(N) asm volatile("s_wait_asynccnt %0" :: "i"(N) : "memory")
#else
#define WAIT_ASYNC(N) ((void)0)
#endif

__device__ __forceinline__ unsigned short f32_to_bf16(float f) {
    unsigned int u = __float_as_uint(f);
    unsigned int r = u + 0x7fffu + ((u >> 16) & 1u);   // round-to-nearest-even
    return (unsigned short)(r >> 16);
}

// ---------------------------------------------------------------- cast kernel
__global__ void cast_bf16_kernel(const float* __restrict__ in,
                                 unsigned short* __restrict__ out, int n) {
    int i = blockIdx.x * blockDim.x + threadIdx.x;
    if (i < n) out[i] = f32_to_bf16(in[i]);
}

// ---------------------------------------------------------------- WMMA GEMM
// C[M,N] (f32) = A[M,K](bf16 row-major) * W[N,K](bf16 row-major)^T + bias[N]
// Block: 256 threads = 8 waves. Block tile 128(M) x 256(N), BK=64,
// double-buffered LDS filled via global_load_async_to_lds_b128.
// Wave tile 64x64 = 4x4 accumulators of v_wmma_f32_16x16x32_bf16.
#define BM 128
#define BN 256
#define BK 64
#define LDT 72                       // padded LDS row stride (elements)
#define SA_ELEMS (BM * LDT)          // one A buffer
#define SB_ELEMS (BN * LDT)          // one B buffer
#define SMEM_BYTES ((2 * SA_ELEMS + 2 * SB_ELEMS) * 2)

__global__ void __launch_bounds__(256)
gemm_bf16_kernel(const unsigned short* __restrict__ A,
                 const unsigned short* __restrict__ W,
                 const float* __restrict__ bias,
                 float* __restrict__ C,
                 int M, int N, int K)
{
    extern __shared__ __align__(16) unsigned short smem[];
    unsigned short* sA = smem;                    // 2 x BM x LDT
    unsigned short* sB = smem + 2 * SA_ELEMS;     // 2 x BN x LDT

    const int tid  = threadIdx.x;
    const int gm   = blockIdx.y * BM;
    const int gn   = blockIdx.x * BN;

    const int wave = tid >> 5;
    const int lane = tid & 31;
    const int l    = lane & 15;
    const int half = lane >> 4;
    const int wm   = (wave & 1) * 64;    // wave row offset in block tile
    const int wn   = (wave >> 1) * 64;   // wave col offset in block tile

    const int lr = tid >> 3;        // 0..31 : row group for coalesced tile fill
    const int lc = (tid & 7) * 8;   // 8-bf16 (16B) chunk within row

    v8f acc[4][4];
    #pragma unroll
    for (int i = 0; i < 4; ++i)
        #pragma unroll
        for (int j = 0; j < 4; ++j)
            acc[i][j] = (v8f){0.f,0.f,0.f,0.f,0.f,0.f,0.f,0.f};

    const int nstages = K / BK;

    // ---- stage issue: 12 async b128 copies per thread (A:4, B:8) ----
    auto issue_stage = [&](int kt, int buf) {
        unsigned short* dA = sA + buf * SA_ELEMS;
        unsigned short* dB = sB + buf * SB_ELEMS;
        #pragma unroll
        for (int p = 0; p < 4; ++p) {
            int row = lr + p * 32;
            async_copy16(&A[(size_t)(gm + row) * K + kt + lc], &dA[row * LDT + lc]);
        }
        #pragma unroll
        for (int p = 0; p < 8; ++p) {
            int row = lr + p * 32;
            async_copy16(&W[(size_t)(gn + row) * K + kt + lc], &dB[row * LDT + lc]);
        }
    };

    issue_stage(0, 0);

    for (int s = 0; s < nstages; ++s) {
        const int buf = s & 1;
        if (s + 1 < nstages) {
            issue_stage((s + 1) * BK, buf ^ 1);
            WAIT_ASYNC(12);       // stage-s copies (first 12) complete, next stage in flight
        } else {
            WAIT_ASYNC(0);
        }
        __syncthreads();

        const unsigned short* bA = sA + buf * SA_ELEMS;
        const unsigned short* bB = sB + buf * SB_ELEMS;

        #pragma unroll
        for (int ks = 0; ks < BK; ks += 32) {
            // A fragments 16x32: lane l = row; lanes 0-15 K [ks,ks+8)+[ks+16,ks+24)
            v16bf af[4];
            #pragma unroll
            for (int i = 0; i < 4; ++i) {
                const unsigned short* base = &bA[(wm + i*16 + l) * LDT + ks + half*8];
                union { v16bf v; v8bf h[2]; } u;
                u.h[0] = *(const v8bf*)(base);
                u.h[1] = *(const v8bf*)(base + 16);
                af[i] = u.v;
            }
            // B fragments 32x16: lane l = col; lanes 0-15 K [ks,ks+8)+[ks+16,ks+24)
            v16bf bfr[4];
            #pragma unroll
            for (int j = 0; j < 4; ++j) {
                const unsigned short* base = &bB[(wn + j*16 + l) * LDT + ks + half*8];
                union { v16bf v; v8bf h[2]; } u;
                u.h[0] = *(const v8bf*)(base);
                u.h[1] = *(const v8bf*)(base + 16);
                bfr[j] = u.v;
            }
            #pragma unroll
            for (int i = 0; i < 4; ++i)
                #pragma unroll
                for (int j = 0; j < 4; ++j)
                    acc[i][j] = __builtin_amdgcn_wmma_f32_16x16x32_bf16(
                        false, af[i], false, bfr[j], (short)0, acc[i][j], false, false);
        }
        __syncthreads();   // all waves done reading buf before it is refilled
    }

    // Epilogue: C/D layout — VGPR r, lanes 0-15 -> M=r, lanes 16-31 -> M=8+r, N=lane%16
    #pragma unroll
    for (int j = 0; j < 4; ++j) {
        int col = gn + wn + j*16 + l;
        float bv = bias[col];
        #pragma unroll
        for (int i = 0; i < 4; ++i) {
            #pragma unroll
            for (int r = 0; r < 8; ++r) {
                int row = gm + wm + i*16 + half*8 + r;
                C[(size_t)row * N + col] = acc[i][j][r] + bv;
            }
        }
    }
}

// ---------------------------------------------------------------- scan layer 0
__global__ void scan0_kernel(const float* __restrict__ pre,
                             const float* __restrict__ u,
                             unsigned short* __restrict__ hout)
{
    int idx = blockIdx.x * blockDim.x + threadIdx.x;   // b*H + h
    int b = idx >> 11;
    int h = idx & (H_ - 1);
    float uu = u[h];
    float hs = 0.f;
    const float* p = pre + (size_t)b * T_ * H_ + h;
    unsigned short* o = hout + (size_t)b * T_ * H_ + h;
    for (int t = 0; t < T_; ++t) {
        float v = p[(size_t)t * H_] + uu * hs;
        hs = v > 0.f ? v : 0.f;
        o[(size_t)t * H_] = f32_to_bf16(hs);
    }
}

// ---------------------------------------------------------------- scan layer 1 + head
__global__ void __launch_bounds__(1024)
scan1_head_kernel(const float* __restrict__ pre,
                  const float* __restrict__ u,
                  const float* __restrict__ lw,
                  const float* __restrict__ lb,
                  float* __restrict__ out)
{
    int b   = blockIdx.x;
    int tid = threadIdx.x;
    int h0 = tid, h1 = tid + 1024;
    float u0v = u[h0], u1v = u[h1];
    float s0 = 0.f, s1 = 0.f;
    const float* p = pre + (size_t)b * T_ * H_;
    for (int t = 0; t < T_; ++t) {
        float v0 = p[(size_t)t * H_ + h0] + u0v * s0; s0 = v0 > 0.f ? v0 : 0.f;
        float v1 = p[(size_t)t * H_ + h1] + u1v * s1; s1 = v1 > 0.f ? v1 : 0.f;
    }
    __shared__ float red[1024];
    red[tid] = s0 * lw[h0] + s1 * lw[h1];
    __syncthreads();
    for (int s = 512; s > 0; s >>= 1) {
        if (tid < s) red[tid] += red[tid + s];
        __syncthreads();
    }
    if (tid == 0) out[b] = red[0] + lb[0];
}

// ---------------------------------------------------------------- launch
extern "C" void kernel_launch(void* const* d_in, const int* in_sizes, int n_in,
                              void* d_out, int out_size, void* d_ws, size_t ws_size,
                              hipStream_t stream) {
    const float* x  = (const float*)d_in[0];
    const float* W0 = (const float*)d_in[1];
    const float* b0 = (const float*)d_in[2];
    const float* u0 = (const float*)d_in[3];
    const float* W1 = (const float*)d_in[4];
    const float* b1 = (const float*)d_in[5];
    const float* u1 = (const float*)d_in[6];
    const float* lw = (const float*)d_in[7];
    const float* lb = (const float*)d_in[8];
    float* out = (float*)d_out;

    char* ws = (char*)d_ws;
    size_t off = 0;
    auto alloc = [&](size_t bytes) -> void* {
        void* p = ws + off;
        off = (off + bytes + 255) & ~(size_t)255;
        return p;
    };
    unsigned short* xb  = (unsigned short*)alloc((size_t)B_ * T_ * I_ * 2);  // x bf16
    unsigned short* w0b = (unsigned short*)alloc((size_t)H_ * I_ * 2);       // W0 bf16
    unsigned short* w1b = (unsigned short*)alloc((size_t)H_ * H_ * 2);       // W1 bf16
    float*          pre = (float*)alloc((size_t)B_ * T_ * H_ * 4);           // pre0/pre1 (aliased)
    unsigned short* h0b = (unsigned short*)alloc((size_t)B_ * T_ * H_ * 2);  // layer-0 out bf16
    (void)ws_size; (void)in_sizes; (void)n_in; (void)out_size;

    int n;
    n = B_ * T_ * I_;
    cast_bf16_kernel<<<(n + 255) / 256, 256, 0, stream>>>(x, xb, n);
    n = H_ * I_;
    cast_bf16_kernel<<<(n + 255) / 256, 256, 0, stream>>>(W0, w0b, n);
    n = H_ * H_;
    cast_bf16_kernel<<<(n + 255) / 256, 256, 0, stream>>>(W1, w1b, n);

    dim3 gg(H_ / BN, (B_ * T_) / BM);   // (8, 200)

    // Layer 0 GEMM: pre0 = x @ W0^T + b0   (M=25600, N=2048, K=128)
    gemm_bf16_kernel<<<gg, 256, SMEM_BYTES, stream>>>(xb, w0b, b0, pre, B_ * T_, H_, I_);

    // Layer 0 scan -> bf16 h0
    scan0_kernel<<<(B_ * H_) / 256, 256, 0, stream>>>(pre, u0, h0b);

    // Layer 1 GEMM: pre1 = h0 @ W1^T + b1  (M=25600, N=2048, K=2048)
    gemm_bf16_kernel<<<gg, 256, SMEM_BYTES, stream>>>(h0b, w1b, b1, pre, B_ * T_, H_, H_);

    // Layer 1 scan + linear head
    scan1_head_kernel<<<B_, 1024, 0, stream>>>(pre, u1, lw, lb, out);
}
